// Spectral_Net_17128329576810
// MI455X (gfx1250) — matrix operand order repaired and checked by reading, not compiled, not run on hardware
//
#include <hip/hip_runtime.h>
#include <hip/hip_bf16.h>

// ---------- types ----------
typedef __attribute__((ext_vector_type(16))) __bf16 v16bf;
typedef __attribute__((ext_vector_type(8)))  float  v8f;

union ABFrag { v16bf v; unsigned u[8]; };

static __device__ __forceinline__ unsigned short f2bf(float f) {
    union { float f; unsigned u; } x; x.f = f;
    unsigned r = x.u + 0x7FFFu + ((x.u >> 16) & 1u);   // round-to-nearest-even
    return (unsigned short)(r >> 16);
}

// A-fragment dword offsets within a 32-K chunk (lane-half folded into base):
// VGPR j<4 -> K=2j (+8*lh), j>=4 -> K=16+2(j-4) (+8*lh)  => q in {0,1,2,3,8,9,10,11}
#define QA(j) (((j) < 4) ? (j) : ((j) + 4))

// ============================================================================
// Kernel A: gather + conv1 + conv2 + conv3  (one block per batch)
// Emits conv3 activations (relu'd) as bf16 twice:
//   Pe[b][ic][p] = x3[b][ic][p]      p=0..79  (even-phase pairs)
//   Po[b][ic][p] = x3[b][ic][p+1]    p=0..79  (odd-phase pairs, x3[80]=0)
// so any window start t reads aligned dword pairs from Pe (t even) / Po (t odd).
// ============================================================================
__global__ __launch_bounds__(256) void front_kernel(
    const float* __restrict__ img, const int* __restrict__ idx_i,
    const int* __restrict__ idx_j,
    const float* __restrict__ w1, const float* __restrict__ b1,
    const float* __restrict__ w2, const float* __restrict__ b2,
    const float* __restrict__ w3, const float* __restrict__ b3,
    unsigned short* __restrict__ Pe, unsigned short* __restrict__ Po)
{
    const int b   = blockIdx.x;
    const int tid = threadIdx.x;

    __shared__ float vecP[82];        // index p+1, halo zeros
    __shared__ float x1P[32][82];
    __shared__ float x2P[64][82];

    if (tid < 82) vecP[tid] = 0.f;
    if (tid < 32) { x1P[tid][0] = 0.f; x1P[tid][81] = 0.f; }
    if (tid < 64) { x2P[tid][0] = 0.f; x2P[tid][81] = 0.f; }
    __syncthreads();

    // gather 68 values: p = k*4 + c
    if (tid < 68) {
        int k = tid >> 2, c = tid & 3;
        int n = (k == 16) ? 3 : (k >> 2);
        int m = (k == 16) ? 4 : (k & 3);
        int di = (n & 1) ? 2 : 0;
        int dj = (n >= 2) ? 128 : 0;
        int ii = idx_i[(b * 4 + n) * 5 + m] + di;
        int jj = idx_j[(b * 4 + n) * 5 + m] + dj;
        vecP[tid + 1] = img[((b * 4 + c) * 256 + ii) * 256 + jj];
    }
    __syncthreads();

    // conv1: 1 -> 32, kw=3, pad=1, relu
    for (int i = tid; i < 32 * 80; i += 256) {
        int oc = i / 80, p = i % 80;
        float s = b1[oc];
        #pragma unroll
        for (int k = 0; k < 3; ++k) s += w1[oc * 3 + k] * vecP[p + k];
        x1P[oc][p + 1] = fmaxf(s, 0.f);
    }
    __syncthreads();

    // conv2: 32 -> 64, kw=3, pad=1, relu
    for (int i = tid; i < 64 * 80; i += 256) {
        int oc = i / 80, p = i % 80;
        float s = b2[oc];
        for (int ic = 0; ic < 32; ++ic) {
            const float* w = w2 + (oc * 32 + ic) * 3;
            #pragma unroll
            for (int k = 0; k < 3; ++k) s += w[k] * x1P[ic][p + k];
        }
        x2P[oc][p + 1] = fmaxf(s, 0.f);
    }
    __syncthreads();

    // conv3: 64 -> 256, kw=3, pad=1, relu -> bf16 stores (two phases)
    for (int i = tid; i < 256 * 80; i += 256) {
        int oc = i / 80, p = i % 80;
        float s = b3[oc];
        for (int ic = 0; ic < 64; ++ic) {
            const float* w = w3 + (oc * 64 + ic) * 3;
            #pragma unroll
            for (int k = 0; k < 3; ++k) s += w[k] * x2P[ic][p + k];
        }
        unsigned short bv = f2bf(fmaxf(s, 0.f));
        unsigned base = ((unsigned)b * 256u + (unsigned)oc) * 80u;
        Pe[base + p] = bv;
        if (p >= 1)  Po[base + p - 1] = bv;
        if (p == 79) Po[base + 79] = 0;       // x3[..][80] = 0
    }
}

// ============================================================================
// Weight pack: dst[q][n] = pack_bf16(src[n][2q], src[n][2q+1]), dst is [K/2][N]
// (q varies fastest across threads -> coalesced reads of src rows)
// ============================================================================
__global__ __launch_bounds__(256) void pack_wt_kernel(
    const float* __restrict__ src, unsigned* __restrict__ dst, int K, int N)
{
    int id = blockIdx.x * blockDim.x + threadIdx.x;
    int half = K / 2;
    if (id >= half * N) return;
    int n = id / half, q = id % half;
    unsigned lo = f2bf(src[n * K + 2 * q]);
    unsigned hi = f2bf(src[n * K + 2 * q + 1]);
    dst[q * N + n] = lo | (hi << 16);
}

// ============================================================================
// Kernel B: conv4 as GEMM  M=(b,t) N=oc K=16384, WMMA bf16, fused bias+relu+sum_t
// grid = 16 b-tiles x 16 oc-tiles (256 blocks), 128 threads (4 waves).
// Wave w owns t = 4w..4w+3 (whole K) and a quarter of K for t=16.
// Fully uniform control flow: EXEC stays all-ones for every WMMA.
// ============================================================================
__global__ __launch_bounds__(128) void conv4_gemm_kernel(
    const unsigned short* __restrict__ Pe, const unsigned short* __restrict__ Po,
    const unsigned* __restrict__ w4pk,     // [8192][256] dword-packed bf16 pairs
    const float* __restrict__ b4,
    unsigned short* __restrict__ sBf)      // [256][256] bf16 row-major
{
    const int lane = threadIdx.x & 31;
    const int wv   = __builtin_amdgcn_readfirstlane(threadIdx.x >> 5);  // scalar 0..3
    const int b0   = (blockIdx.x & 15) * 16;
    const int oc0  = (blockIdx.x >> 4) * 16;
    const int row  = lane & 15;
    const int lh   = lane >> 4;

    v8f acc[5];   // acc[0..3]: t = 4*wv + i (full K); acc[4]: t=16, K quarter
    #pragma unroll
    for (int i = 0; i < 5; ++i)
        #pragma unroll
        for (int e = 0; e < 8; ++e) acc[i][e] = 0.f;

    const unsigned* PeD = (const unsigned*)Pe;
    const unsigned* PoD = (const unsigned*)Po;

    // per-lane bases with lane-half and wave t-shift folded in
    const unsigned rb0 = (unsigned)(b0 + row) * 256u * 40u + (unsigned)(lh * 4);
    const unsigned* peB  = PeD + rb0 + 2 * wv;   // t = 4wv, 4wv+2 (q, q+1)
    const unsigned* poB  = PoD + rb0 + 2 * wv;   // t = 4wv+1, 4wv+3
    const unsigned* pe16 = PeD + rb0 + 8;        // t = 16
    const unsigned* bB   = w4pk + (unsigned)(oc0 + row) + (unsigned)(lh * 4 * 256);

    for (int seg = 0; seg < 4; ++seg) {          // K segments of 64 ic
        const bool do16 = (seg == wv);           // scalar condition
        for (int icl = 0; icl < 64; ++icl) {
            const int ic = seg * 64 + icl;
            const unsigned ico = (unsigned)ic * 40u;
            const unsigned* pe = peB + ico;
            const unsigned* po = poB + ico;
            const unsigned* bw = bB + (unsigned)ic * 8192u;   // ic*32 rows * 256

            #pragma unroll
            for (int half = 0; half < 2; ++half) {
                // B fragment: immediate offsets off one base
                ABFrag bf;
                #pragma unroll
                for (int j = 0; j < 8; ++j)
                    bf.u[j] = bw[(half * 16 + QA(j)) * 256];

                // 4 full-t fragments: two bases (Pe/Po), offsets q and q+1
                #pragma unroll
                for (int i = 0; i < 4; ++i) {
                    const unsigned* p = (i & 1) ? po : pe;
                    const unsigned  d = (unsigned)(i >> 1) + (unsigned)(half * 16);
                    ABFrag af;
                    #pragma unroll
                    for (int j = 0; j < 8; ++j) af.u[j] = p[d + QA(j)];
                    acc[i] = __builtin_amdgcn_wmma_f32_16x16x32_bf16(
                        false, af.v, false, bf.v, (short)0, acc[i], false, false);
                }

                // t = 16: this wave's K quarter only (scalar branch)
                if (do16) {
                    ABFrag af;
                    #pragma unroll
                    for (int j = 0; j < 8; ++j)
                        af.u[j] = pe16[ico + (unsigned)(half * 16) + QA(j)];
                    acc[4] = __builtin_amdgcn_wmma_f32_16x16x32_bf16(
                        false, af.v, false, bf.v, (short)0, acc[4], false, false);
                }
            }
        }
    }

    // epilogue: +b4, relu, sum over this wave's 4 t's; t=16 reduced pre-relu
    const float bn = b4[oc0 + row];
    __shared__ float red[4][8][32];
    __shared__ float red16[4][8][32];
    #pragma unroll
    for (int e = 0; e < 8; ++e) {
        float s = 0.f;
        #pragma unroll
        for (int i = 0; i < 4; ++i) s += fmaxf(acc[i][e] + bn, 0.f);
        red[wv][e][lane]   = s;
        red16[wv][e][lane] = acc[4][e];
    }
    __syncthreads();

    for (int el = threadIdx.x; el < 256; el += 128) {
        int e = el >> 5, L = el & 31;
        float s   = red[0][e][L] + red[1][e][L] + red[2][e][L] + red[3][e][L];
        float y16 = red16[0][e][L] + red16[1][e][L] + red16[2][e][L] + red16[3][e][L];
        float bnl = b4[oc0 + (L & 15)];
        s += fmaxf(y16 + bnl, 0.f);
        int M = e + 8 * (L >> 4);     // C/D layout: VGPR e, lane-half
        int N = L & 15;
        sBf[(b0 + M) * 256 + oc0 + N] = f2bf(s);
    }
}

// ============================================================================
// Kernel C: dense layer out = relu(in @ W^T + bias), 256x256x256, WMMA bf16.
// One wave per 16x16 tile, 8 K-steps. Writes bf16 (chain) and/or f32 (final).
// ============================================================================
__global__ __launch_bounds__(32) void fc_layer_kernel(
    const unsigned short* __restrict__ inBf,   // [256][256] bf16 row-major
    const unsigned* __restrict__ wPk,          // [128][256] packed
    const float* __restrict__ bias,
    unsigned short* __restrict__ outBf,        // may be null
    float* __restrict__ outF)                  // may be null
{
    const int lane = threadIdx.x;
    const int b0 = (blockIdx.x & 15) * 16;
    const int o0 = (blockIdx.x >> 4) * 16;
    const int row = lane & 15;
    const int lh  = lane >> 4;

    const unsigned* pA = (const unsigned*)inBf + (unsigned)(b0 + row) * 128u + lh * 4;
    const unsigned* pB = wPk + (unsigned)(o0 + row) + (unsigned)(lh * 4 * 256);

    v8f acc;
    #pragma unroll
    for (int e = 0; e < 8; ++e) acc[e] = 0.f;

    #pragma unroll
    for (int ks = 0; ks < 8; ++ks) {
        ABFrag af, bf;
        #pragma unroll
        for (int j = 0; j < 8; ++j) {
            af.u[j] = pA[ks * 16 + QA(j)];
            bf.u[j] = pB[(ks * 16 + QA(j)) * 256];
        }
        acc = __builtin_amdgcn_wmma_f32_16x16x32_bf16(
            false, af.v, false, bf.v, (short)0, acc, false, false);
    }

    const float bn = bias[o0 + row];
    #pragma unroll
    for (int e = 0; e < 8; ++e) {
        float v = fmaxf(acc[e] + bn, 0.f);
        int M = e + 8 * lh, N = row;
        int idx = (b0 + M) * 256 + o0 + N;
        if (outBf) outBf[idx] = f2bf(v);
        if (outF)  outF[idx]  = v;
    }
}

// ============================================================================
// launch
// ============================================================================
extern "C" void kernel_launch(void* const* d_in, const int* in_sizes, int n_in,
                              void* d_out, int out_size, void* d_ws, size_t ws_size,
                              hipStream_t stream) {
    (void)in_sizes; (void)n_in; (void)out_size; (void)ws_size;
    const float* img   = (const float*)d_in[0];
    const int*   idx_i = (const int*)  d_in[1];
    const int*   idx_j = (const int*)  d_in[2];
    const float* w1 = (const float*)d_in[3];  const float* b1 = (const float*)d_in[4];
    const float* w2 = (const float*)d_in[5];  const float* b2 = (const float*)d_in[6];
    const float* w3 = (const float*)d_in[7];  const float* b3 = (const float*)d_in[8];
    const float* w4 = (const float*)d_in[9];  const float* b4 = (const float*)d_in[10];
    const float* lw1 = (const float*)d_in[11]; const float* lb1 = (const float*)d_in[12];
    const float* lw2 = (const float*)d_in[13]; const float* lb2 = (const float*)d_in[14];

    char* ws = (char*)d_ws;
    unsigned short* Pe    = (unsigned short*)(ws);                  // 10,485,760 B
    unsigned short* Po    = (unsigned short*)(ws + 10485760);       // 10,485,760 B
    unsigned*       w4pk  = (unsigned*)      (ws + 20971520);       //  8,388,608 B
    unsigned*       lw1pk = (unsigned*)      (ws + 29360128);       //    131,072 B
    unsigned*       lw2pk = (unsigned*)      (ws + 29491200);       //    131,072 B
    unsigned short* sBf   = (unsigned short*)(ws + 29622272);       //    131,072 B
    unsigned short* h1Bf  = (unsigned short*)(ws + 29753344);       //    131,072 B

    // pack weights into WMMA-B layout (bf16 K-pair dwords)
    pack_wt_kernel<<<(8192 * 256 + 255) / 256, 256, 0, stream>>>(w4,  w4pk, 16384, 256);
    pack_wt_kernel<<<( 128 * 256 + 255) / 256, 256, 0, stream>>>(lw1, lw1pk,  256, 256);
    pack_wt_kernel<<<( 128 * 256 + 255) / 256, 256, 0, stream>>>(lw2, lw2pk,  256, 256);

    // gather + conv1/2/3
    front_kernel<<<256, 256, 0, stream>>>(img, idx_i, idx_j,
                                          w1, b1, w2, b2, w3, b3, Pe, Po);

    // conv4 GEMM (36.5 GFLOP) + bias + relu + sum_t
    conv4_gemm_kernel<<<256, 128, 0, stream>>>(Pe, Po, w4pk, b4, sBf);

    // two FC layers
    fc_layer_kernel<<<256, 32, 0, stream>>>(sBf,  lw1pk, lb1, h1Bf, nullptr);
    fc_layer_kernel<<<256, 32, 0, stream>>>(h1Bf, lw2pk, lb2, nullptr, (float*)d_out);
}
